// KAE_29661044146816
// MI455X (gfx1250) — matrix operand order repaired and checked by reading, not compiled, not run on hardware
//
#include <hip/hip_runtime.h>

// ---------------------------------------------------------------------------
// k-sparse autoencoder forward for MI455X (gfx1250, wave32, WMMA bf16)
//   h  = X (4096x12288) . W^T (12288x4096)      [bf16 WMMA, f32 accum]
//   hm = h masked to entries strictly > 65th-largest per row (top-64 kept)
//   Xr = sigmoid(hm . W + bias)                 [bf16 WMMA, f32 accum]
// CDNA5 paths: v_wmma_f32_16x16x32_bf16, tensor_load_to_lds (TDM, TENSORcnt,
// LDS pad-on-load), global_load_async_to_lds_b128 (ASYNCcnt), global_prefetch,
// split barriers.
// ---------------------------------------------------------------------------

typedef __bf16 bf16_t;
typedef __bf16 v4bf  __attribute__((ext_vector_type(4)));
typedef __bf16 v8bf  __attribute__((ext_vector_type(8)));
typedef __bf16 v16bf __attribute__((ext_vector_type(16)));
typedef float  v8f   __attribute__((ext_vector_type(8)));
typedef unsigned v4u __attribute__((ext_vector_type(4)));
typedef unsigned v8u __attribute__((ext_vector_type(8)));

#define B_DIM  4096      // batch
#define N_HID  4096      // hidden units
#define D_DIM  12288     // 64*64*3
#define K_KEEP 64        // active units

#define BM  128          // rows of C per workgroup
#define BN  128          // cols of C per workgroup
#define BK  32           // K step (matches 16x16x32 bf16 WMMA)
#define PAD 8            // bf16 row pad: 80B stride -> conflict-free, 16B aligned
#define LDA (BK + PAD)

// ---- CDNA5 async global->LDS copy (ASYNCcnt) ------------------------------
// GVS mode: mem_addr = SADDR(u64) + VADDR(i32); LDS byte address in VDST vgpr.
// Generic LDS pointers carry the LDS offset in their low 32 bits (ISA 10.2).
__device__ __forceinline__ void async_copy_b128(const void* gbase,
                                                unsigned voff_bytes,
                                                void* lds_ptr) {
    asm volatile("global_load_async_to_lds_b128 %0, %1, %2"
                 :: "v"((unsigned)(unsigned long long)lds_ptr),
                    "v"(voff_bytes),
                    "s"(gbase)
                 : "memory");
}
__device__ __forceinline__ void wait_async() {
    asm volatile("s_wait_asynccnt 0x0" ::: "memory");
}

// ---- CDNA5 Tensor Data Mover: 2D tile global->LDS (TENSORcnt) -------------
// Builds D# groups 0/1 (ISA 08, §8.3/8.4) in SGPRs and issues the 2-group
// form of tensor_load_to_lds (VADDR2/3 = NULL, tensors up to 2D).
// tensor dims == tile dims -> no OOB; optional LDS pad-on-load inserts
// pad_amount DWORDs every pad_interval DWORDs (used to keep the padded,
// bank-conflict-free LDS row stride).
__device__ __forceinline__ void tdm_load_2d(unsigned lds_addr, const void* gaddr,
                                            unsigned tile_d0, unsigned tile_d1,
                                            unsigned long long row_stride_elems,
                                            unsigned data_size_code,
                                            unsigned pad_interval_code,
                                            unsigned pad_amount_code,
                                            bool pad_en)
{
    union { unsigned w[4]; v4u v; } g0;
    union { unsigned w[8]; v8u v; } g1;
    const unsigned long long ga = (unsigned long long)gaddr;
    const unsigned td1 = tile_d1 ? tile_d1 : 1;

    g0.w[0] = 1u;                                     // count=1, user descriptor
    g0.w[1] = lds_addr;                               // LDS byte address
    g0.w[2] = (unsigned)(ga & 0xffffffffu);           // global_addr[31:0]
    g0.w[3] = (unsigned)((ga >> 32) & 0x01ffffffu)    // global_addr[56:32]
            | (2u << 30);                             // type = 2 ("image")

    g1.w[0] = (data_size_code << 16)                  // 0=1B 1=2B 2=4B 3=8B
            | ((pad_en ? 1u : 0u) << 20)
            | (pad_interval_code << 22)               // 3 -> every 16 DWORDs
            | (pad_amount_code << 25);                // 3 -> insert 4 DWORDs
    g1.w[1] = (tile_d0 & 0xffffu) << 16;              // tensor_dim0[15:0]
    g1.w[2] = (tile_d0 >> 16) | ((td1 & 0xffffu) << 16);   // td0 hi | tensor_dim1 lo
    g1.w[3] = (td1 >> 16) | ((tile_d0 & 0xffffu) << 16);   // td1 hi | tile_dim0
    g1.w[4] = (tile_d1 & 0xffffu);                    // tile_dim1 (tile_dim2=0)
    g1.w[5] = (unsigned)(row_stride_elems & 0xffffffffu);  // tensor_dim0_stride lo
    g1.w[6] = (unsigned)((row_stride_elems >> 32) & 0xffffu); // stride hi
    g1.w[7] = 0;                                      // tensor_dim1_stride (unused)

    asm volatile("tensor_load_to_lds %0, %1"
                 :: "s"(g0.v), "s"(g1.v)
                 : "memory");
}

// ------------------------- encode: h = X . W^T -----------------------------
// Both operands are K-contiguous (row-major, K = D_DIM); stage tiles into LDS
// with f32 -> bf16 conversion in flight (must stay on the VALU path).
__global__ __launch_bounds__(256) void kae_encode_gemm(
    const float* __restrict__ X, const float* __restrict__ W,
    float* __restrict__ H)
{
    __shared__ bf16_t As[2][BM][LDA];   // 20 KB
    __shared__ bf16_t Bs[2][BN][LDA];   // 20 KB

    const int tid  = threadIdx.x;
    const int lane = tid & 31;
    const int wave = tid >> 5;          // 8 waves: 4 (M) x 2 (N)
    const int wm   = (wave >> 1) * 32;
    const int wn   = (wave & 1) * 64;
    const long m0  = (long)blockIdx.x * BM;
    const long n0  = (long)blockIdx.y * BN;

    v8f acc[2][4] = {};

    auto stage = [&](int k0, int buf) {
        #pragma unroll
        for (int i = 0; i < 4; ++i) {                 // A: 128x32 f32
            int q = tid + i * 256;                    // 1024 float4 chunks
            int r = q >> 3;
            int c = (q & 7) << 2;
            const float* gp = X + (m0 + r) * D_DIM + k0 + c;
            const float4 f = *(const float4*)gp;
            if (k0 + BK < D_DIM) __builtin_prefetch(gp + BK, 0, 0);
            v4bf p = { (bf16_t)f.x, (bf16_t)f.y, (bf16_t)f.z, (bf16_t)f.w };
            *(v4bf*)&As[buf][r][c] = p;
        }
        #pragma unroll
        for (int i = 0; i < 4; ++i) {                 // B: 128x32 f32
            int q = tid + i * 256;
            int r = q >> 3;
            int c = (q & 7) << 2;
            const float* gp = W + (n0 + r) * D_DIM + k0 + c;
            const float4 f = *(const float4*)gp;
            if (k0 + BK < D_DIM) __builtin_prefetch(gp + BK, 0, 0);
            v4bf p = { (bf16_t)f.x, (bf16_t)f.y, (bf16_t)f.z, (bf16_t)f.w };
            *(v4bf*)&Bs[buf][r][c] = p;
        }
    };

    stage(0, 0);
    __syncthreads();

    const int mlane = lane & 15;
    const int kbA   = (lane >> 4) << 3;   // A frag: half-waves interleave K by 8
    const int kbB   = (lane >> 4) << 3;   // B frag: half-waves interleave K by 8

    for (int k0 = 0; k0 < D_DIM; k0 += BK) {
        const int buf = (k0 / BK) & 1;
        if (k0 + BK < D_DIM) stage(k0 + BK, buf ^ 1);

        v16bf a[2], b[4];
        #pragma unroll
        for (int t = 0; t < 2; ++t) {
            const bf16_t* p = &As[buf][wm + t * 16 + mlane][0];
            v8bf lo = *(const v8bf*)(p + kbA);
            v8bf hi = *(const v8bf*)(p + kbA + 16);
            a[t] = __builtin_shufflevector(lo, hi, 0,1,2,3,4,5,6,7,
                                                   8,9,10,11,12,13,14,15);
        }
        #pragma unroll
        for (int u = 0; u < 4; ++u) {
            const bf16_t* p = &Bs[buf][wn + u * 16 + mlane][0];
            v8bf lo = *(const v8bf*)(p + kbB);
            v8bf hi = *(const v8bf*)(p + kbB + 16);
            b[u] = __builtin_shufflevector(lo, hi, 0,1,2,3,4,5,6,7,
                                                   8,9,10,11,12,13,14,15);
        }
        #pragma unroll
        for (int t = 0; t < 2; ++t)
            #pragma unroll
            for (int u = 0; u < 4; ++u)
                acc[t][u] = __builtin_amdgcn_wmma_f32_16x16x32_bf16(
                    false, a[t], false, b[u], (short)0, acc[t][u], false, false);
        __syncthreads();
    }

    const int rsel = (lane >> 4) << 3;   // upper half-wave holds M+8
    #pragma unroll
    for (int t = 0; t < 2; ++t)
        #pragma unroll
        for (int u = 0; u < 4; ++u) {
            long row = m0 + wm + t * 16 + rsel;
            long col = n0 + wn + u * 16 + mlane;
            #pragma unroll
            for (int j = 0; j < 8; ++j)
                H[(row + j) * N_HID + col] = acc[t][u][j];
        }
}

// --------------------- top-k select: exact bit radix -----------------------
// theta = 65th-largest of the row; keep h strictly > theta (top-64 survive).
// Row is async-DMA'd into LDS (raw f32), then transformed in place to an
// order-preserving u32 key; 32-step MSB-first radix threshold build.
__global__ __launch_bounds__(256) void kae_topk(
    const float* __restrict__ H, bf16_t* __restrict__ HM)
{
    __shared__ unsigned us[N_HID];      // 16 KB (f32 bits, then u32 keys)
    __shared__ int cnt;

    const int b   = blockIdx.x;
    const int tid = threadIdx.x;
    const float* row = H + (long)b * N_HID;

    #pragma unroll
    for (int i = 0; i < 4; ++i) {       // 1024 x 16B chunks, async -> LDS
        int q = tid + i * 256;
        async_copy_b128(row, (unsigned)(q * 16), &us[q * 4]);
    }
    wait_async();
    __syncthreads();

    for (int i = tid; i < N_HID; i += 256) {   // in-place key transform
        unsigned v = us[i];
        us[i] = (v >> 31) ? ~v : (v | 0x80000000u);
    }
    __syncthreads();

    unsigned ans = 0;
    for (int bit = 31; bit >= 0; --bit) {
        const unsigned cand = ans | (1u << bit);
        if (tid == 0) cnt = 0;
        __syncthreads();
        int c = 0;
        for (int i = tid; i < N_HID; i += 256) c += (us[i] >= cand);
        #pragma unroll
        for (int off = 16; off > 0; off >>= 1) c += __shfl_down(c, off, 32);
        if ((tid & 31) == 0) atomicAdd(&cnt, c);
        __syncthreads();
        if (cnt >= K_KEEP + 1) ans = cand;   // invariant: count(>= ans) >= 65
        __syncthreads();
    }
    // ans == 65th-largest key; strictly-greater keys are the top-64
    for (int i = tid; i < N_HID; i += 256) {
        const unsigned u = us[i];
        float f = 0.0f;
        if (u > ans) {
            unsigned vb = (u & 0x80000000u) ? (u & 0x7fffffffu) : ~u;
            f = __uint_as_float(vb);
        }
        HM[(long)b * N_HID + i] = (bf16_t)f;
    }
}

// ------------------ decode: out = sigmoid(hm . W + bias) -------------------
// A (hm) is bf16, K-contiguous: one TDM tensor_load_to_lds per stage moves the
// whole 128x(64B) tile, inserting 16B LDS padding per row to keep the
// bank-conflict-free stride. Issued by wave 0 only via a *scalar* branch
// (tensor ops ignore EXEC, so a masked vector branch would still issue).
// B (W) is consumed K-major -> transpose during f32->bf16 staging.
__global__ __launch_bounds__(256) void kae_decode_gemm(
    const bf16_t* __restrict__ HM, const float* __restrict__ W,
    const float* __restrict__ bias, float* __restrict__ OUT)
{
    __shared__ bf16_t As[2][BM][LDA];
    __shared__ bf16_t Bs[2][BN][LDA];

    const int tid  = threadIdx.x;
    const int lane = tid & 31;
    const int wave = tid >> 5;
    const int wm   = (wave >> 1) * 32;
    const int wn   = (wave & 1) * 64;
    const long m0  = (long)blockIdx.x * BM;   // batch rows
    const long n0  = (long)blockIdx.y * BN;   // output-feature cols
    const bool tdm_wave = (__builtin_amdgcn_readfirstlane(wave) == 0);

    v8f acc[2][4] = {};

    auto stage = [&](int k0, int buf) {
        if (tdm_wave)                                  // A: 128x32 bf16 via TDM
            tdm_load_2d((unsigned)(unsigned long long)&As[buf][0][0],
                        HM + m0 * N_HID + k0,
                        /*tile_d0=*/BK, /*tile_d1=*/BM,
                        /*row_stride=*/N_HID,
                        /*data_size=*/1,               // 2-byte elements
                        /*pad_interval=*/3,            // every 16 DWORDs (64B row)
                        /*pad_amount=*/3,              // +4 DWORDs (16B pad)
                        /*pad_en=*/true);
        #pragma unroll
        for (int i = 0; i < 4; ++i) {                 // B: 32(k) x 128(d) f32, transpose
            int q  = tid + i * 256;                   // 1024 float4 chunks
            int kr = q >> 5;
            int dc = (q & 31) << 2;
            const float* gp = W + (long)(k0 + kr) * D_DIM + n0 + dc;
            const float4 f = *(const float4*)gp;
            if (k0 + BK < N_HID) __builtin_prefetch(gp + (long)BK * D_DIM, 0, 0);
            Bs[buf][dc + 0][kr] = (bf16_t)f.x;
            Bs[buf][dc + 1][kr] = (bf16_t)f.y;
            Bs[buf][dc + 2][kr] = (bf16_t)f.z;
            Bs[buf][dc + 3][kr] = (bf16_t)f.w;
        }
    };

    stage(0, 0);
    if (tdm_wave) __builtin_amdgcn_s_wait_tensorcnt(0);
    __syncthreads();

    const int mlane = lane & 15;
    const int kbA   = (lane >> 4) << 3;
    const int kbB   = (lane >> 4) << 3;

    for (int k0 = 0; k0 < N_HID; k0 += BK) {
        const int buf = (k0 / BK) & 1;
        const bool more = (k0 + BK < N_HID);
        if (more) stage(k0 + BK, buf ^ 1);

        v16bf a[2], b[4];
        #pragma unroll
        for (int t = 0; t < 2; ++t) {
            const bf16_t* p = &As[buf][wm + t * 16 + mlane][0];
            v8bf lo = *(const v8bf*)(p + kbA);
            v8bf hi = *(const v8bf*)(p + kbA + 16);
            a[t] = __builtin_shufflevector(lo, hi, 0,1,2,3,4,5,6,7,
                                                   8,9,10,11,12,13,14,15);
        }
        #pragma unroll
        for (int u = 0; u < 4; ++u) {
            const bf16_t* p = &Bs[buf][wn + u * 16 + mlane][0];
            v8bf lo = *(const v8bf*)(p + kbB);
            v8bf hi = *(const v8bf*)(p + kbB + 16);
            b[u] = __builtin_shufflevector(lo, hi, 0,1,2,3,4,5,6,7,
                                                   8,9,10,11,12,13,14,15);
        }
        #pragma unroll
        for (int t = 0; t < 2; ++t)
            #pragma unroll
            for (int u = 0; u < 4; ++u)
                acc[t][u] = __builtin_amdgcn_wmma_f32_16x16x32_bf16(
                    false, a[t], false, b[u], (short)0, acc[t][u], false, false);
        if (more && tdm_wave) __builtin_amdgcn_s_wait_tensorcnt(0); // overlapped DMA done
        __syncthreads();
    }

    const int rsel = (lane >> 4) << 3;
    #pragma unroll
    for (int t = 0; t < 2; ++t)
        #pragma unroll
        for (int u = 0; u < 4; ++u) {
            long row = m0 + wm + t * 16 + rsel;
            long col = n0 + wn + u * 16 + mlane;
            const float bc = bias[col];
            #pragma unroll
            for (int j = 0; j < 8; ++j) {
                float x = acc[t][u][j] + bc;
                OUT[(row + j) * D_DIM + col] = 1.0f / (1.0f + __expf(-x));
            }
        }
}

// ------------------------------- launcher ----------------------------------
extern "C" void kernel_launch(void* const* d_in, const int* in_sizes, int n_in,
                              void* d_out, int out_size, void* d_ws, size_t ws_size,
                              hipStream_t stream)
{
    (void)in_sizes; (void)n_in; (void)out_size; (void)ws_size;
    const float* X    = (const float*)d_in[0];
    const float* W    = (const float*)d_in[1];
    const float* bias = (const float*)d_in[2];
    float*       out  = (float*)d_out;

    float*  H  = (float*)d_ws;                                          // 64 MB
    bf16_t* HM = (bf16_t*)((char*)d_ws + (size_t)B_DIM * N_HID * 4);    // +32 MB

    dim3 blk(256);
    kae_encode_gemm<<<dim3(B_DIM / BM, N_HID / BN), blk, 0, stream>>>(X, W, H);
    kae_topk<<<B_DIM, blk, 0, stream>>>(H, HM);
    kae_decode_gemm<<<dim3(B_DIM / BM, D_DIM / BN), blk, 0, stream>>>(HM, W, bias, out);
}
